// UTransformerBlock_65025804861928
// MI455X (gfx1250) — matrix operand order repaired and checked by reading, not compiled, not run on hardware
//
#include <hip/hip_runtime.h>

typedef unsigned short u16;
typedef __bf16 bf16_t;
typedef __attribute__((ext_vector_type(16))) bf16_t v16bf;
typedef __attribute__((ext_vector_type(8)))  float  v8f;
typedef __attribute__((ext_vector_type(4)))  int    v4i_ty;

#define BQ 4
#define SQ 1024
#define DQ 1024
#define EQ 4
#define HIDQ 2816
#define RQ (BQ*SQ)      // 4096 rows
#define SIXD (6*DQ)     // 6144

// ---------------- CDNA5 async global->LDS staging ----------------
#if __has_builtin(__builtin_amdgcn_global_load_async_to_lds_b128)
#define ASYNC_OK 1
#else
#define ASYNC_OK 0
#endif

typedef __attribute__((address_space(1))) v4i_ty g_v4i;
typedef __attribute__((address_space(3))) v4i_ty l_v4i;

__device__ __forceinline__ void wait_async0() {
#if ASYNC_OK
#if __has_builtin(__builtin_amdgcn_s_wait_asynccnt)
    __builtin_amdgcn_s_wait_asynccnt(0);
#else
    asm volatile("s_wait_asynccnt 0x0" ::: "memory");
#endif
#endif
}

// copy 16 bf16 (32B) global -> LDS for this thread
__device__ __forceinline__ void copy16_g2l(const u16* g, u16* l) {
#if ASYNC_OK
    g_v4i* gp = (g_v4i*)(g);
    l_v4i* lp = (l_v4i*)(l);
    __builtin_amdgcn_global_load_async_to_lds_b128(gp,     lp,     0, 0);
    __builtin_amdgcn_global_load_async_to_lds_b128(gp + 1, lp + 1, 0, 0);
#else
    const uint4* s = (const uint4*)g;
    uint4 d0 = s[0], d1 = s[1];
    ((uint4*)l)[0] = d0;
    ((uint4*)l)[1] = d1;
#endif
}

// ---------------- small helpers ----------------
__device__ __forceinline__ u16 f2bf(float f) {
    unsigned u = __builtin_bit_cast(unsigned, f);
    unsigned r = (u + 0x7FFFu + ((u >> 16) & 1u)) >> 16;
    return (u16)r;
}
__device__ __forceinline__ float bf2f(u16 u) {
    unsigned v = ((unsigned)u) << 16;
    return __builtin_bit_cast(float, v);
}

__device__ __forceinline__ float block_reduce_sum(float v, float* sbuf) {
    int t = threadIdx.x;
    sbuf[t] = v; __syncthreads();
    for (int s = 128; s > 0; s >>= 1) { if (t < s) sbuf[t] += sbuf[t + s]; __syncthreads(); }
    float r = sbuf[0]; __syncthreads();
    return r;
}
__device__ __forceinline__ float block_reduce_max(float v, float* sbuf) {
    int t = threadIdx.x;
    sbuf[t] = v; __syncthreads();
    for (int s = 128; s > 0; s >>= 1) { if (t < s) sbuf[t] = fmaxf(sbuf[t], sbuf[t + s]); __syncthreads(); }
    float r = sbuf[0]; __syncthreads();
    return r;
}

// ---------------- bf16 WMMA GEMM, double-buffered LDS pipeline ----------------
// C[M,N] = op(A) * op(B); A,B bf16; C f32 or bf16.
// TA=false: A row-major [M,K]. TA=true: A stored [K,M] (A^T used), lda = row stride.
// TB=false: B row-major [K,N]. TB=true: B stored [N,K] (B^T used).
#define BM 128
#define BN 128
#define KT 32

union FragBF { uint4 q[2]; v16bf v; };

template<bool TA, bool TB, bool OUT_BF16>
__global__ __launch_bounds__(256) void gemm_bf16(
    const u16* __restrict__ A, const u16* __restrict__ B, void* __restrict__ C,
    int M, int N, int K, int lda, int ldb, int ldc)
{
    __shared__ __align__(16) u16 As[2][BM][KT];   // [buf][m][k]
    __shared__ __align__(16) u16 Bs[2][KT][BN];   // [buf][k][n]

    const int tid  = threadIdx.x;
    const int bm   = blockIdx.y * BM;
    const int bn   = blockIdx.x * BN;
    const int wid  = tid >> 5;
    const int lane = tid & 31;
    const int wm   = (wid >> 2) * 64;   // 0 / 64
    const int wn   = (wid & 3) * 32;    // 0,32,64,96

    const v8f vzero = {0.f,0.f,0.f,0.f,0.f,0.f,0.f,0.f};
    v8f acc[4][2];
#pragma unroll
    for (int i = 0; i < 4; ++i)
#pragma unroll
        for (int j = 0; j < 2; ++j) acc[i][j] = vzero;

    const int m_frag = lane & 15;
    const int kb     = (lane >> 4) * 8;   // A frag K base: 0 or 8

    auto stage = [&](int k0, int buf) {
        if (!TA) {
            const int r = tid >> 1;
            const int h = (tid & 1) * 16;
            copy16_g2l(A + (size_t)(bm + r) * lda + k0 + h, &As[buf][r][h]);
        } else {
            const int kk = tid >> 3;
            const int mc = (tid & 7) * 16;
            const u16* src = A + (size_t)(k0 + kk) * lda + bm + mc;
            __builtin_prefetch(src, 0, 1);
#pragma unroll
            for (int i = 0; i < 16; ++i) As[buf][mc + i][kk] = src[i];
        }
        if (!TB) {
            const int kk = tid >> 3;
            const int nc = (tid & 7) * 16;
            copy16_g2l(B + (size_t)(k0 + kk) * ldb + bn + nc, &Bs[buf][kk][nc]);
        } else {
            const int nn = tid >> 1;
            const int h  = (tid & 1) * 16;
            const u16* src = B + (size_t)(bn + nn) * ldb + k0 + h;
            __builtin_prefetch(src, 0, 1);
#pragma unroll
            for (int i = 0; i < 16; ++i) Bs[buf][h + i][nn] = src[i];
        }
    };

    // prologue: fill buffer 0
    stage(0, 0);
    wait_async0();
    __syncthreads();

    int cur = 0;
    for (int k0 = 0; k0 < K; k0 += KT) {
        // issue fill of the other buffer while computing on this one
        if (k0 + KT < K) stage(k0 + KT, cur ^ 1);

        FragBF bfr[2];
#pragma unroll
        for (int j = 0; j < 2; ++j) {
            const int n0 = wn + j * 16;
#pragma unroll
            for (int s = 0; s < 16; ++s)
                bfr[j].v[s] = __builtin_bit_cast(bf16_t, Bs[cur][kb + (s & 7) + ((s >> 3) << 4)][n0 + m_frag]);
        }
#pragma unroll
        for (int i = 0; i < 4; ++i) {
            FragBF af;
            const int m0 = wm + i * 16 + m_frag;
            af.q[0] = *(const uint4*)&As[cur][m0][kb];
            af.q[1] = *(const uint4*)&As[cur][m0][kb + 16];
#pragma unroll
            for (int j = 0; j < 2; ++j) {
                acc[i][j] = __builtin_amdgcn_wmma_f32_16x16x32_bf16(
                    false, af.v, false, bfr[j].v, (short)0, acc[i][j], false, false);
            }
        }

        wait_async0();
        __syncthreads();
        cur ^= 1;
    }

    // ---- store C ----
#pragma unroll
    for (int i = 0; i < 4; ++i) {
#pragma unroll
        for (int j = 0; j < 2; ++j) {
            const int col  = bn + wn + j * 16 + (lane & 15);
            const int row0 = bm + wm + i * 16 + ((lane >> 4) << 3);
#pragma unroll
            for (int r = 0; r < 8; ++r) {
                float v = acc[i][j][r];
                if (OUT_BF16) ((u16*)C)[(size_t)(row0 + r) * ldc + col] = f2bf(v);
                else          ((float*)C)[(size_t)(row0 + r) * ldc + col] = v;
            }
        }
    }
}

// ---------------- elementwise / reduction kernels ----------------
__global__ __launch_bounds__(256) void cvt_bf16_kernel(const float* __restrict__ src,
                                                       u16* __restrict__ dst, int n)
{
    for (int i = blockIdx.x * 256 + threadIdx.x; i < n; i += gridDim.x * 256)
        dst[i] = f2bf(src[i]);
}

// ada = silu(cond) @ W + b   (cond [4,1024], W [1024,6144]); one block per column
__global__ __launch_bounds__(256) void ada_kernel(const float* __restrict__ cond,
                                                  const float* __restrict__ W,
                                                  const float* __restrict__ bias,
                                                  float* __restrict__ ada)
{
    __shared__ float sbuf[256];
    const int c = blockIdx.x;
    float acc[BQ] = {0.f, 0.f, 0.f, 0.f};
    for (int k = threadIdx.x; k < DQ; k += 256) {
        float wv = W[(size_t)k * SIXD + c];
#pragma unroll
        for (int b = 0; b < BQ; ++b) {
            float a = cond[b * DQ + k];
            float s = a / (1.f + __expf(-a));
            acc[b] += s * wv;
        }
    }
#pragma unroll
    for (int b = 0; b < BQ; ++b) {
        float tot = block_reduce_sum(acc[b], sbuf);
        if (threadIdx.x == 0) ada[(size_t)b * SIXD + c] = tot + bias[c];
    }
}

// LayerNorm over D=1024, optional adaLN modulate, bf16 out. One block per row.
__global__ __launch_bounds__(256) void ln_mod_kernel(const float* __restrict__ x,
                                                     const float* __restrict__ w,
                                                     const float* __restrict__ bias,
                                                     const float* __restrict__ ada,
                                                     int shift_chunk, int scale_chunk,
                                                     u16* __restrict__ out, float eps)
{
    __shared__ float sbuf[256];
    const int row = blockIdx.x;
    const int b   = row >> 10;
    const int t   = threadIdx.x;
    const float* xr = x + (size_t)row * DQ;
    float r[4];
#pragma unroll
    for (int i = 0; i < 4; ++i) r[i] = xr[t + i * 256];
    float mu = block_reduce_sum(r[0] + r[1] + r[2] + r[3], sbuf) * (1.f / DQ);
    float sq = 0.f;
#pragma unroll
    for (int i = 0; i < 4; ++i) { float d = r[i] - mu; sq += d * d; }
    float var  = block_reduce_sum(sq, sbuf) * (1.f / DQ);
    float rstd = rsqrtf(var + eps);
#pragma unroll
    for (int i = 0; i < 4; ++i) {
        int c = t + i * 256;
        float y = (r[i] - mu) * rstd * w[c] + bias[c];
        if (ada) {
            float sc = ada[(size_t)b * SIXD + scale_chunk * DQ + c];
            float sh = ada[(size_t)b * SIXD + shift_chunk * DQ + c];
            y = y * (1.f + sc) + sh;
        }
        out[(size_t)row * DQ + c] = f2bf(y);
    }
}

// row softmax over 1024, f32 in -> bf16 out; one block per row
__global__ __launch_bounds__(256) void softmax_kernel(const float* __restrict__ x,
                                                      u16* __restrict__ out)
{
    __shared__ float sbuf[256];
    const int row = blockIdx.x;
    const int t   = threadIdx.x;
    const float* xr = x + (size_t)row * DQ;
    float r[4];
#pragma unroll
    for (int i = 0; i < 4; ++i) r[i] = xr[t + i * 256];
    float m = fmaxf(fmaxf(r[0], r[1]), fmaxf(r[2], r[3]));
    m = block_reduce_max(m, sbuf);
    float e[4]; float s = 0.f;
#pragma unroll
    for (int i = 0; i < 4; ++i) { e[i] = __expf(r[i] - m); s += e[i]; }
    float inv = 1.f / block_reduce_sum(s, sbuf);
#pragma unroll
    for (int i = 0; i < 4; ++i) out[(size_t)row * DQ + t + i * 256] = f2bf(e[i] * inv);
}

// logits[row][e] = h(bf16)[row] . rw[:,e] + rb[e]; one block per row
__global__ __launch_bounds__(256) void router_kernel(const u16* __restrict__ h,
                                                     const float* __restrict__ rw,
                                                     const float* __restrict__ rb,
                                                     float* __restrict__ logits)
{
    __shared__ float sbuf[256];
    const int row = blockIdx.x;
    float acc[EQ] = {0.f, 0.f, 0.f, 0.f};
    for (int k = threadIdx.x; k < DQ; k += 256) {
        float hv = bf2f(h[(size_t)row * DQ + k]);
#pragma unroll
        for (int e = 0; e < EQ; ++e) acc[e] += hv * rw[(size_t)k * EQ + e];
    }
#pragma unroll
    for (int e = 0; e < EQ; ++e) {
        float tot = block_reduce_sum(acc[e], sbuf);
        if (threadIdx.x == 0) logits[(size_t)row * EQ + e] = tot + rb[e];
    }
}

// L2 norm of logits along the SEQUENCE dim, per (b,e). grid = 16 blocks.
__global__ __launch_bounds__(256) void rnorm_kernel(const float* __restrict__ logits,
                                                    float* __restrict__ nrm)
{
    __shared__ float sbuf[256];
    const int b = blockIdx.x >> 2, e = blockIdx.x & 3;
    float acc = 0.f;
    for (int s = threadIdx.x; s < SQ; s += 256) {
        float v = logits[(size_t)((b << 10) + s) * EQ + e];
        acc += v * v;
    }
    float tot = block_reduce_sum(acc, sbuf);
    if (threadIdx.x == 0) nrm[blockIdx.x] = fmaxf(sqrtf(tot), 1e-12f);
}

// softmax over E, top-2 combine weights. 16 blocks x 256 = 4096 rows.
__global__ __launch_bounds__(256) void probs_kernel(const float* __restrict__ logits,
                                                    const float* __restrict__ nrm,
                                                    float* __restrict__ probs,
                                                    float* __restrict__ comb)
{
    const int r = blockIdx.x * 256 + threadIdx.x;
    const int b = r >> 10;
    float l[EQ], m = -3.4e38f;
#pragma unroll
    for (int e = 0; e < EQ; ++e) { l[e] = logits[(size_t)r * EQ + e] / nrm[b * EQ + e]; m = fmaxf(m, l[e]); }
    float p[EQ], s = 0.f;
#pragma unroll
    for (int e = 0; e < EQ; ++e) { p[e] = __expf(l[e] - m); s += p[e]; }
    float inv = 1.f / s;
#pragma unroll
    for (int e = 0; e < EQ; ++e) p[e] *= inv;
    int i0 = 0;
#pragma unroll
    for (int e = 1; e < EQ; ++e) if (p[e] > p[i0]) i0 = e;
    int i1 = -1;
#pragma unroll
    for (int e = 0; e < EQ; ++e) if (e != i0 && (i1 < 0 || p[e] > p[i1])) i1 = e;
#pragma unroll
    for (int e = 0; e < EQ; ++e) {
        probs[(size_t)r * EQ + e] = p[e];
        comb[(size_t)r * EQ + e]  = (e == i0 || e == i1) ? p[e] : 0.f;
    }
}

// aux = sum_{s,e} (1/E - mean_b probs)^2  -> out[0]
__global__ __launch_bounds__(256) void aux_kernel(const float* __restrict__ probs,
                                                  float* __restrict__ out_aux)
{
    __shared__ float sbuf[256];
    float acc = 0.f;
    for (int i = threadIdx.x; i < SQ * EQ; i += 256) {
        int s = i >> 2, e = i & 3;
        float avg = 0.f;
#pragma unroll
        for (int b = 0; b < BQ; ++b) avg += probs[(size_t)(((b << 10) + s) << 2) + e];
        avg *= 0.25f;
        float d = 0.25f - avg;
        acc += d * d;
    }
    float tot = block_reduce_sum(acc, sbuf);
    if (threadIdx.x == 0) out_aux[0] = tot;
}

// act = bf16( sin(h1) * h3 )
__global__ __launch_bounds__(256) void act_kernel(const float* __restrict__ h1,
                                                  const float* __restrict__ h3,
                                                  u16* __restrict__ act, int n)
{
    for (int i = blockIdx.x * 256 + threadIdx.x; i < n; i += gridDim.x * 256)
        act[i] = f2bf(sinf(h1[i]) * h3[i]);
}

// x1 = x + gate_msa[b,d] * proj   (gate = ada chunk 2)
__global__ __launch_bounds__(256) void residual_gate_kernel(const float* __restrict__ x,
                                                            const float* __restrict__ proj,
                                                            const float* __restrict__ ada,
                                                            float* __restrict__ out, int n)
{
    for (int i = blockIdx.x * 256 + threadIdx.x; i < n; i += gridDim.x * 256) {
        int d = i & (DQ - 1);
        int b = i >> 20;      // i / (S*D)
        float g = ada[(size_t)b * SIXD + 2 * DQ + d];
        out[i] = x[i] + g * proj[i];
    }
}

// out (+)= comb[row,e] * oute
__global__ __launch_bounds__(256) void combine_kernel(const float* __restrict__ oute,
                                                      const float* __restrict__ comb,
                                                      float* __restrict__ out,
                                                      int e, int init, int n)
{
    for (int i = blockIdx.x * 256 + threadIdx.x; i < n; i += gridDim.x * 256) {
        int r = i >> 10;
        float w = comb[(size_t)(r << 2) + e];
        float v = w * oute[i];
        if (init) out[i] = v; else out[i] += v;
    }
}

// ---------------- host orchestration ----------------
extern "C" void kernel_launch(void* const* d_in, const int* in_sizes, int n_in,
                              void* d_out, int out_size, void* d_ws, size_t ws_size,
                              hipStream_t stream)
{
    (void)in_sizes; (void)n_in; (void)out_size; (void)ws_size;
    const float* x     = (const float*)d_in[0];
    const float* adaln = (const float*)d_in[2];
    const float* wq    = (const float*)d_in[3];
    const float* wk    = (const float*)d_in[4];
    const float* wv    = (const float*)d_in[5];
    const float* wo    = (const float*)d_in[6];
    const float* qn_w  = (const float*)d_in[7];
    const float* qn_b  = (const float*)d_in[8];
    const float* kn_w  = (const float*)d_in[9];
    const float* kn_b  = (const float*)d_in[10];
    const float* an_w  = (const float*)d_in[11];
    const float* an_b  = (const float*)d_in[12];
    const float* fn_w  = (const float*)d_in[13];
    const float* fn_b  = (const float*)d_in[14];
    const float* w1    = (const float*)d_in[15];
    const float* w2    = (const float*)d_in[16];
    const float* w3    = (const float*)d_in[17];
    const float* rw    = (const float*)d_in[18];
    const float* rb    = (const float*)d_in[19];
    const float* aw    = (const float*)d_in[20];
    const float* ab    = (const float*)d_in[21];
    float* out = (float*)d_out;

    char* wsb = (char*)d_ws;
    size_t off = 0;
    auto alloc = [&](size_t bytes) -> void* {
        void* p = wsb + off;
        off += (bytes + 255) & ~(size_t)255;
        return p;
    };
    const size_t DD  = (size_t)DQ * DQ;            // 1M
    const size_t RD  = (size_t)RQ * DQ;            // 4M
    const size_t EDH = (size_t)EQ * DQ * HIDQ;     // 11.5M
    const size_t RH  = (size_t)RQ * HIDQ;          // 11.5M

    u16*  wq_bf = (u16*)alloc(DD * 2);
    u16*  wk_bf = (u16*)alloc(DD * 2);
    u16*  wv_bf = (u16*)alloc(DD * 2);
    u16*  wo_bf = (u16*)alloc(DD * 2);
    u16*  w1_bf = (u16*)alloc(EDH * 2);
    u16*  w2_bf = (u16*)alloc(EDH * 2);
    u16*  w3_bf = (u16*)alloc(EDH * 2);
    float* ada  = (float*)alloc((size_t)BQ * SIXD * 4);
    u16*  hb    = (u16*)alloc(RD * 2);
    float* qpre = (float*)alloc(RD * 4);
    float* kpre = (float*)alloc(RD * 4);
    u16*  qb    = (u16*)alloc(RD * 2);
    u16*  kb_   = (u16*)alloc(RD * 2);
    u16*  vb    = (u16*)alloc(RD * 2);
    float* sc   = (float*)alloc(RD * 4);           // scores [B,D,D] (B*D = R rows)
    u16*  attnb = (u16*)alloc(RD * 2);
    u16*  attos = (u16*)alloc(RD * 2);
    float* proj = (float*)alloc(RD * 4);
    float* x1   = (float*)alloc(RD * 4);
    u16*  h2    = (u16*)alloc(RD * 2);
    float* logits = (float*)alloc((size_t)RQ * EQ * 4);
    float* nrm    = (float*)alloc(16 * 4);
    float* probs  = (float*)alloc((size_t)RQ * EQ * 4);
    float* comb   = (float*)alloc((size_t)RQ * EQ * 4);
    float* h1buf  = (float*)alloc(RH * 4);
    float* h3buf  = (float*)alloc(RH * 4);
    u16*  actb   = (u16*)alloc(RH * 2);
    float* oute   = (float*)alloc(RD * 4);

    const dim3 blk(256);
    const int EG = 2048;  // elementwise grid

    // weight conversions f32 -> bf16
    cvt_bf16_kernel<<<EG, blk, 0, stream>>>(wq, wq_bf, (int)DD);
    cvt_bf16_kernel<<<EG, blk, 0, stream>>>(wk, wk_bf, (int)DD);
    cvt_bf16_kernel<<<EG, blk, 0, stream>>>(wv, wv_bf, (int)DD);
    cvt_bf16_kernel<<<EG, blk, 0, stream>>>(wo, wo_bf, (int)DD);
    cvt_bf16_kernel<<<EG, blk, 0, stream>>>(w1, w1_bf, (int)EDH);
    cvt_bf16_kernel<<<EG, blk, 0, stream>>>(w2, w2_bf, (int)EDH);
    cvt_bf16_kernel<<<EG, blk, 0, stream>>>(w3, w3_bf, (int)EDH);

    // adaLN modulation vector
    ada_kernel<<<SIXD, blk, 0, stream>>>(adaln, aw, ab, ada);

    // h = modulate(ln(x)) -> bf16
    ln_mod_kernel<<<RQ, blk, 0, stream>>>(x, an_w, an_b, ada, 0, 1, hb, 1e-5f);

    // q/k/v projections
    gemm_bf16<false,false,false><<<dim3(DQ/BN, RQ/BM), blk, 0, stream>>>(hb, wq_bf, qpre, RQ, DQ, DQ, DQ, DQ, DQ);
    gemm_bf16<false,false,false><<<dim3(DQ/BN, RQ/BM), blk, 0, stream>>>(hb, wk_bf, kpre, RQ, DQ, DQ, DQ, DQ, DQ);
    gemm_bf16<false,false,true ><<<dim3(DQ/BN, RQ/BM), blk, 0, stream>>>(hb, wv_bf, vb,   RQ, DQ, DQ, DQ, DQ, DQ);

    // q_norm / k_norm
    ln_mod_kernel<<<RQ, blk, 0, stream>>>(qpre, qn_w, qn_b, nullptr, 0, 0, qb, 1e-5f);
    ln_mod_kernel<<<RQ, blk, 0, stream>>>(kpre, kn_w, kn_b, nullptr, 0, 0, kb_, 1e-5f);

    // scores[b] = q_b^T @ k_b   (feature-dim attention, [D,D])
    for (int b = 0; b < BQ; ++b) {
        gemm_bf16<true,false,false><<<dim3(DQ/BN, DQ/BM), blk, 0, stream>>>(
            qb + (size_t)b * SQ * DQ, kb_ + (size_t)b * SQ * DQ, sc + (size_t)b * DQ * DQ,
            DQ, DQ, SQ, DQ, DQ, DQ);
    }
    softmax_kernel<<<RQ, blk, 0, stream>>>(sc, attnb);

    // out_s[b] = v_b @ attn_b^T
    for (int b = 0; b < BQ; ++b) {
        gemm_bf16<false,true,true><<<dim3(DQ/BN, SQ/BM), blk, 0, stream>>>(
            vb + (size_t)b * SQ * DQ, attnb + (size_t)b * DQ * DQ, attos + (size_t)b * SQ * DQ,
            SQ, DQ, DQ, DQ, DQ, DQ);
    }

    // proj = out_s @ wo ; x1 = x + gate_msa * proj
    gemm_bf16<false,false,false><<<dim3(DQ/BN, RQ/BM), blk, 0, stream>>>(attos, wo_bf, proj, RQ, DQ, DQ, DQ, DQ, DQ);
    residual_gate_kernel<<<EG, blk, 0, stream>>>(x, proj, ada, x1, (int)RD);

    // h2 = modulate(ln(x1)) for MLP
    ln_mod_kernel<<<RQ, blk, 0, stream>>>(x1, fn_w, fn_b, ada, 3, 4, h2, 1e-5f);

    // router -> seq-dim L2 normalize -> softmax/top2
    router_kernel<<<RQ, blk, 0, stream>>>(h2, rw, rb, logits);
    rnorm_kernel<<<16, blk, 0, stream>>>(logits, nrm);
    probs_kernel<<<16, blk, 0, stream>>>(logits, nrm, probs, comb);

    // experts (dense, sequential, reusing buffers)
    for (int e = 0; e < EQ; ++e) {
        const u16* w1e = w1_bf + (size_t)e * DQ * HIDQ;
        const u16* w3e = w3_bf + (size_t)e * DQ * HIDQ;
        const u16* w2e = w2_bf + (size_t)e * HIDQ * DQ;
        gemm_bf16<false,false,false><<<dim3(HIDQ/BN, RQ/BM), blk, 0, stream>>>(h2, w1e, h1buf, RQ, HIDQ, DQ, DQ, HIDQ, HIDQ);
        gemm_bf16<false,false,false><<<dim3(HIDQ/BN, RQ/BM), blk, 0, stream>>>(h2, w3e, h3buf, RQ, HIDQ, DQ, DQ, HIDQ, HIDQ);
        act_kernel<<<EG, blk, 0, stream>>>(h1buf, h3buf, actb, (int)RH);
        gemm_bf16<false,false,false><<<dim3(DQ/BN, RQ/BM), blk, 0, stream>>>(actb, w2e, oute, RQ, DQ, HIDQ, HIDQ, DQ, DQ);
        combine_kernel<<<EG, blk, 0, stream>>>(oute, comb, out, e, (e == 0) ? 1 : 0, (int)RD);
    }

    // aux loss scalar appended after x
    aux_kernel<<<1, blk, 0, stream>>>(probs, out + RD);
}
